// multi_head_self_attention_layer_90117003805169
// MI455X (gfx1250) — matrix operand order repaired and checked by reading, not compiled, not run on hardware
//
#include <hip/hip_runtime.h>
#include <hip/hip_bf16.h>

// ---------------- problem constants ----------------
#define BB   4
#define NN   2048
#define DD   512
#define HH   8
#define DHH  64
#define EPSLN 1e-5f

typedef _Float16 h16;
typedef __attribute__((ext_vector_type(16))) _Float16 v16h;
typedef __attribute__((ext_vector_type(8)))  _Float16 v8h;
typedef __attribute__((ext_vector_type(8)))  float    v8f;

union V16 { v16h v; v8h h[2]; };

__device__ __forceinline__ v8f wmma32(v16h a, v16h b, v8f c) {
  // D = A(16x32 f16) * B(32x16 f16) + C(16x16 f32)
  return __builtin_amdgcn_wmma_f32_16x16x32_f16(
      /*neg_a=*/false, a, /*neg_b=*/false, b,
      /*c_mod=*/(short)0, c, /*reuse_a=*/false, /*reuse_b=*/false);
}

// ---------------- LayerNorm stage 1: partial sums of x ----------------
__global__ __launch_bounds__(256)
void ln_partial_kernel(const float* __restrict__ x, float* __restrict__ part) {
  const int b = blockIdx.y, g = blockIdx.x, G = gridDim.x;
  const int CH = (NN * DD) / G;                       // 16384 for G=64
  const float* p = x + (size_t)b * NN * DD + (size_t)g * CH;
  float s = 0.f, ss = 0.f;
  for (int i = threadIdx.x; i < CH; i += blockDim.x) { float v = p[i]; s += v; ss += v * v; }
  __shared__ float sh[2][256];
  sh[0][threadIdx.x] = s; sh[1][threadIdx.x] = ss; __syncthreads();
  for (int o = 128; o > 0; o >>= 1) {
    if (threadIdx.x < o) { sh[0][threadIdx.x] += sh[0][threadIdx.x + o];
                           sh[1][threadIdx.x] += sh[1][threadIdx.x + o]; }
    __syncthreads();
  }
  if (threadIdx.x == 0) {
    part[(b * G + g) * 2 + 0] = sh[0][0];
    part[(b * G + g) * 2 + 1] = sh[1][0];
  }
}

// ---------------- finalize mean / rsqrt(var+eps) per batch ----------------
__global__ __launch_bounds__(32)
void ln_finalize_kernel(const float* __restrict__ part, float* __restrict__ stats,
                        int G, float invCnt) {
  const int b = blockIdx.x;
  float s = 0.f, ss = 0.f;
  for (int g = threadIdx.x; g < G; g += 32) { s += part[(b * G + g) * 2]; ss += part[(b * G + g) * 2 + 1]; }
  for (int m = 16; m >= 1; m >>= 1) { s += __shfl_xor(s, m, 32); ss += __shfl_xor(ss, m, 32); }
  if (threadIdx.x == 0) {
    float mu = s * invCnt;
    float var = ss * invCnt - mu * mu;
    stats[b * 2 + 0] = mu;
    stats[b * 2 + 1] = rsqrtf(var + EPSLN);
  }
}

// ---------------- xl = LN1(x)*w+b, plus partial sums for LN2 ----------------
__global__ __launch_bounds__(256)
void xl_kernel(const float* __restrict__ x, const float* __restrict__ lnw,
               const float* __restrict__ lnb, const float* __restrict__ stats1,
               float* __restrict__ xl, float* __restrict__ part2) {
  const int b = blockIdx.y, g = blockIdx.x, G = gridDim.x;
  const int CH = (NN * DD) / G;
  const float mu = stats1[b * 2], rs = stats1[b * 2 + 1];
  const int base = g * CH;
  const float* xp = x + (size_t)b * NN * DD;
  float* xlp = xl + (size_t)b * NN * DD;
  float s = 0.f, ss = 0.f;
  for (int i = threadIdx.x; i < CH; i += blockDim.x) {
    int nd = base + i;
    float v = (xp[nd] - mu) * rs * lnw[nd] + lnb[nd];
    xlp[nd] = v;
    s += v; ss += v * v;
  }
  __shared__ float sh[2][256];
  sh[0][threadIdx.x] = s; sh[1][threadIdx.x] = ss; __syncthreads();
  for (int o = 128; o > 0; o >>= 1) {
    if (threadIdx.x < o) { sh[0][threadIdx.x] += sh[0][threadIdx.x + o];
                           sh[1][threadIdx.x] += sh[1][threadIdx.x + o]; }
    __syncthreads();
  }
  if (threadIdx.x == 0) {
    part2[(b * G + g) * 2 + 0] = sh[0][0];
    part2[(b * G + g) * 2 + 1] = sh[1][0];
  }
}

// ---------------- xi = LN2(xl), cast to f16 ----------------
__global__ __launch_bounds__(256)
void xi_kernel(const float* __restrict__ xl, const float* __restrict__ stats2,
               h16* __restrict__ xi) {
  const int i = blockIdx.x * blockDim.x + threadIdx.x;   // exact cover of B*N*D
  const int b = i / (NN * DD);
  const float mu = stats2[b * 2], rs = stats2[b * 2 + 1];
  xi[i] = (h16)((xl[i] - mu) * rs);
}

// ---------------- weight transpose + f16 cast: wT[d][c] = w[c][d] ----------------
__global__ __launch_bounds__(256)
void wcvt_kernel(const float* __restrict__ wq, const float* __restrict__ wk,
                 const float* __restrict__ wv, const float* __restrict__ wo,
                 h16* __restrict__ wqT, h16* __restrict__ wkT,
                 h16* __restrict__ wvT, h16* __restrict__ woT) {
  const int i = blockIdx.x * blockDim.x + threadIdx.x;   // over DD*DD
  const int d = i / DD, c = i % DD;
  const int src = c * DD + d;   // [H,DH,D] flattens to [512][512] with col index h*64+dh
  wqT[i] = (h16)wq[src];
  wkT[i] = (h16)wk[src];
  wvT[i] = (h16)wv[src];
  woT[i] = (h16)wo[src];
}

// ---------------- QKV projection GEMM (WMMA) ----------------
// grid: (8 heads, 64 row-tiles, 3 {q,k,v}); 256 threads = 8 waves, 128x64 tile
__global__ __launch_bounds__(256)
void qkv_gemm_kernel(const h16* __restrict__ xi,
                     const h16* __restrict__ wqT, const h16* __restrict__ wkT,
                     const h16* __restrict__ wvT,
                     h16* __restrict__ q, h16* __restrict__ k, h16* __restrict__ v) {
  const int head = blockIdx.x;
  const int rowbase = blockIdx.y * 128;
  const int which = blockIdx.z;
  const h16* wT  = (which == 0) ? wqT : (which == 1) ? wkT : wvT;
  h16* outp      = (which == 0) ? q   : (which == 1) ? k   : v;

  __shared__ h16 lds_a[128][40];   // rows x 32 K (pad->80B stride, 16B aligned)
  __shared__ h16 lds_b[64][40];    // cols x 32 K (transposed)

  const int tid = threadIdx.x;
  const int wave = tid >> 5, lane = tid & 31;
  const int lrow = lane & 15, lhalf = lane >> 4;
  const int colbase = head * 64;

  v8f acc[4] = {};

  for (int k0 = 0; k0 < DD; k0 += 32) {
    __syncthreads();
    { // stage A: 128 rows x 32 halfs
      int r = tid >> 1, part = tid & 1;
      const h16* src = xi + (size_t)(rowbase + r) * DD + k0 + part * 16;
      *(v8h*)&lds_a[r][part * 16 + 0] = *(const v8h*)(src + 0);
      *(v8h*)&lds_a[r][part * 16 + 8] = *(const v8h*)(src + 8);
    }
    { // stage B transposed: lds_b[col][K]
      int d = tid & 31, cg = tid >> 5;
      const h16* src = wT + (size_t)(k0 + d) * DD + colbase + cg * 8;
      v8h w8 = *(const v8h*)src;
#pragma unroll
      for (int j = 0; j < 8; j++) lds_b[cg * 8 + j][d] = w8[j];
    }
    __syncthreads();

    V16 af;
    af.h[0] = *(const v8h*)&lds_a[wave * 16 + lrow][lhalf * 8];
    af.h[1] = *(const v8h*)&lds_a[wave * 16 + lrow][16 + lhalf * 8];
#pragma unroll
    for (int f = 0; f < 4; f++) {
      V16 bf;
      bf.h[0] = *(const v8h*)&lds_b[f * 16 + lrow][lhalf * 8];
      bf.h[1] = *(const v8h*)&lds_b[f * 16 + lrow][16 + lhalf * 8];
      acc[f] = wmma32(af.v, bf.v, acc[f]);
    }
  }

  // store as [B,H,N,DH] f16
#pragma unroll
  for (int f = 0; f < 4; f++) {
#pragma unroll
    for (int r = 0; r < 8; r++) {
      int row = rowbase + wave * 16 + r + 8 * lhalf;     // 0..8191 == b*2048+n
      int b = row >> 11, n = row & (NN - 1);
      int dh = f * 16 + lrow;
      outp[((size_t)(b * HH + head) * NN + n) * DHH + dh] = (h16)acc[f][r];
    }
  }
}

// ---------------- Flash attention (WMMA, online softmax) ----------------
// grid: (N/64, H, B); 128 threads = 4 waves; each wave owns 16 query rows
__global__ __launch_bounds__(128)
void attn_kernel(const h16* __restrict__ q, const h16* __restrict__ k,
                 const h16* __restrict__ v, h16* __restrict__ ocat) {
  const int b = blockIdx.z, h = blockIdx.y, rowtile = blockIdx.x;
  const int tid = threadIdx.x, wave = tid >> 5, lane = tid & 31;
  const int lrow = lane & 15, lhalf = lane >> 4;

  __shared__ h16 lds_k[32][72];        // keys x feats (144B stride, conflict-free)
  __shared__ h16 lds_v[64][40];        // feats x keys (transposed)
  __shared__ h16 lds_p[4][16][40];     // per-wave P relayout bounce

  const size_t bh = ((size_t)b * HH + h) * NN;
  const h16* qp = q + bh * DHH;
  const h16* kp = k + bh * DHH;
  const h16* vp = v + bh * DHH;

  // Q A-fragments stay in registers for the whole key sweep
  const int qrow = rowtile * 64 + wave * 16 + lrow;
  V16 qa[2];
#pragma unroll
  for (int s = 0; s < 2; s++) {
    const h16* src = qp + (size_t)qrow * DHH + s * 32;
    qa[s].h[0] = *(const v8h*)(src + lhalf * 8);
    qa[s].h[1] = *(const v8h*)(src + 16 + lhalf * 8);
  }

  float mrow[8], lsum[8];
  v8f acc[4] = {};
#pragma unroll
  for (int r = 0; r < 8; r++) { mrow[r] = -3.0e38f; lsum[r] = 0.f; }

  for (int j0 = 0; j0 < NN; j0 += 32) {
    __syncthreads();
    { // stage K chunk row-major [32 keys][64]
      int key = tid >> 2, part = tid & 3;
      const h16* src = kp + (size_t)(j0 + key) * DHH + part * 16;
      *(v8h*)&lds_k[key][part * 16 + 0] = *(const v8h*)(src + 0);
      *(v8h*)&lds_k[key][part * 16 + 8] = *(const v8h*)(src + 8);
    }
    { // stage V chunk transposed: lds_v[feat][key]
      int key = tid >> 2, g = tid & 3;
      const h16* src = vp + (size_t)(j0 + key) * DHH + g * 16;
      v8h v0 = *(const v8h*)(src + 0);
      v8h v1 = *(const v8h*)(src + 8);
#pragma unroll
      for (int j = 0; j < 8; j++) {
        lds_v[g * 16 + j][key]     = v0[j];
        lds_v[g * 16 + 8 + j][key] = v1[j];
      }
    }
    __syncthreads();

    // S = Q * K^T : two 16x16 tiles over this 32-key chunk
    v8f sfr[2];
#pragma unroll
    for (int t = 0; t < 2; t++) {
      v8f c = {};
#pragma unroll
      for (int s = 0; s < 2; s++) {  // 64 features in two K=32 steps
        V16 bf;
        bf.h[0] = *(const v8h*)&lds_k[t * 16 + lrow][s * 32 + lhalf * 8];
        bf.h[1] = *(const v8h*)&lds_k[t * 16 + lrow][s * 32 + 16 + lhalf * 8];
        c = wmma32(qa[s].v, bf.v, c);
      }
      sfr[t] = c;
    }

    // online softmax (row r lives in VGPR r; lanes 0-15 row r, lanes 16-31 row r+8)
    float alpha[8];
#pragma unroll
    for (int r = 0; r < 8; r++) {
      float mx = fmaxf(sfr[0][r], sfr[1][r]);
#pragma unroll
      for (int m = 8; m >= 1; m >>= 1) mx = fmaxf(mx, __shfl_xor(mx, m, 32));
      float mnew = fmaxf(mrow[r], mx);
      float a  = __expf(mrow[r] - mnew);
      float p0 = __expf(sfr[0][r] - mnew);
      float p1 = __expf(sfr[1][r] - mnew);
      sfr[0][r] = p0; sfr[1][r] = p1;
      float rs = p0 + p1;
#pragma unroll
      for (int m = 8; m >= 1; m >>= 1) rs += __shfl_xor(rs, m, 32);
      lsum[r] = lsum[r] * a + rs;
      mrow[r] = mnew;
      alpha[r] = a;
    }
#pragma unroll
    for (int f = 0; f < 4; f++)
#pragma unroll
      for (int r = 0; r < 8; r++) acc[f][r] *= alpha[r];

    // P: C-layout -> A-layout via per-wave LDS bounce
#pragma unroll
    for (int t = 0; t < 2; t++)
#pragma unroll
      for (int r = 0; r < 8; r++)
        lds_p[wave][r + 8 * lhalf][t * 16 + lrow] = (h16)sfr[t][r];

    V16 pa;
    pa.h[0] = *(const v8h*)&lds_p[wave][lrow][lhalf * 8];
    pa.h[1] = *(const v8h*)&lds_p[wave][lrow][16 + lhalf * 8];

    // O += P * V
#pragma unroll
    for (int f = 0; f < 4; f++) {
      V16 bf;
      bf.h[0] = *(const v8h*)&lds_v[f * 16 + lrow][lhalf * 8];
      bf.h[1] = *(const v8h*)&lds_v[f * 16 + lrow][16 + lhalf * 8];
      acc[f] = wmma32(pa.v, bf.v, acc[f]);
    }
  }

  // finalize: O / l, store concat-heads layout [B,N,D] f16
#pragma unroll
  for (int r = 0; r < 8; r++) {
    float inv = 1.0f / lsum[r];
    int row = rowtile * 64 + wave * 16 + r + 8 * lhalf;
#pragma unroll
    for (int f = 0; f < 4; f++) {
      int feat = f * 16 + lrow;
      ocat[((size_t)b * NN + row) * DD + h * DHH + feat] = (h16)(acc[f][r] * inv);
    }
  }
}

// ---------------- output projection + bias + residual ----------------
// grid: (8 col-tiles, 64 row-tiles); 256 threads = 8 waves; 128x64 tile
__global__ __launch_bounds__(256)
void out_proj_kernel(const h16* __restrict__ ocat, const h16* __restrict__ woT,
                     const float* __restrict__ bo, const float* __restrict__ xl,
                     float* __restrict__ out) {
  const int colbase = blockIdx.x * 64;
  const int rowbase = blockIdx.y * 128;

  __shared__ h16 lds_a[128][40];
  __shared__ h16 lds_b[64][40];

  const int tid = threadIdx.x;
  const int wave = tid >> 5, lane = tid & 31;
  const int lrow = lane & 15, lhalf = lane >> 4;

  v8f acc[4] = {};

  for (int k0 = 0; k0 < DD; k0 += 32) {
    __syncthreads();
    {
      int r = tid >> 1, part = tid & 1;
      const h16* src = ocat + (size_t)(rowbase + r) * DD + k0 + part * 16;
      *(v8h*)&lds_a[r][part * 16 + 0] = *(const v8h*)(src + 0);
      *(v8h*)&lds_a[r][part * 16 + 8] = *(const v8h*)(src + 8);
    }
    {
      int d = tid & 31, cg = tid >> 5;
      const h16* src = woT + (size_t)(k0 + d) * DD + colbase + cg * 8;
      v8h w8 = *(const v8h*)src;
#pragma unroll
      for (int j = 0; j < 8; j++) lds_b[cg * 8 + j][d] = w8[j];
    }
    __syncthreads();

    V16 af;
    af.h[0] = *(const v8h*)&lds_a[wave * 16 + lrow][lhalf * 8];
    af.h[1] = *(const v8h*)&lds_a[wave * 16 + lrow][16 + lhalf * 8];
#pragma unroll
    for (int f = 0; f < 4; f++) {
      V16 bf;
      bf.h[0] = *(const v8h*)&lds_b[f * 16 + lrow][lhalf * 8];
      bf.h[1] = *(const v8h*)&lds_b[f * 16 + lrow][16 + lhalf * 8];
      acc[f] = wmma32(af.v, bf.v, acc[f]);
    }
  }

#pragma unroll
  for (int f = 0; f < 4; f++) {
#pragma unroll
    for (int r = 0; r < 8; r++) {
      int row = rowbase + wave * 16 + r + 8 * lhalf;
      int c = colbase + f * 16 + lrow;
      size_t idx = (size_t)row * DD + c;
      out[idx] = acc[f][r] + bo[c] + xl[idx];
    }
  }
}

// ---------------- host side ----------------
extern "C" void kernel_launch(void* const* d_in, const int* in_sizes, int n_in,
                              void* d_out, int out_size, void* d_ws, size_t ws_size,
                              hipStream_t stream) {
  const float* x   = (const float*)d_in[0];
  const float* lnw = (const float*)d_in[1];
  const float* lnb = (const float*)d_in[2];
  const float* wq  = (const float*)d_in[3];
  const float* wk  = (const float*)d_in[4];
  const float* wv  = (const float*)d_in[5];
  const float* wo  = (const float*)d_in[6];
  const float* bo  = (const float*)d_in[7];
  float* out = (float*)d_out;

  // bump allocator over workspace (~60.8 MB total)
  char* ws = (char*)d_ws;
  auto alloc = [&](size_t bytes) -> char* {
    char* p = ws;
    ws += (bytes + 255) & ~(size_t)255;
    return p;
  };
  const int G = 64;
  float* part1  = (float*)alloc((size_t)BB * G * 2 * 4);
  float* stats1 = (float*)alloc((size_t)BB * 2 * 4);
  float* part2  = (float*)alloc((size_t)BB * G * 2 * 4);
  float* stats2 = (float*)alloc((size_t)BB * 2 * 4);
  float* xl     = (float*)alloc((size_t)BB * NN * DD * 4);
  h16*   xi     = (h16*)  alloc((size_t)BB * NN * DD * 2);
  h16*   wqT    = (h16*)  alloc((size_t)DD * DD * 2);
  h16*   wkT    = (h16*)  alloc((size_t)DD * DD * 2);
  h16*   wvT    = (h16*)  alloc((size_t)DD * DD * 2);
  h16*   woT    = (h16*)  alloc((size_t)DD * DD * 2);
  h16*   qb     = (h16*)  alloc((size_t)BB * HH * NN * DHH * 2);
  h16*   kb     = (h16*)  alloc((size_t)BB * HH * NN * DHH * 2);
  h16*   vb     = (h16*)  alloc((size_t)BB * HH * NN * DHH * 2);
  h16*   ocat   = (h16*)  alloc((size_t)BB * NN * DD * 2);

  const float invCnt = 1.0f / (float)(NN * DD);

  ln_partial_kernel<<<dim3(G, BB), 256, 0, stream>>>(x, part1);
  ln_finalize_kernel<<<BB, 32, 0, stream>>>(part1, stats1, G, invCnt);
  xl_kernel<<<dim3(G, BB), 256, 0, stream>>>(x, lnw, lnb, stats1, xl, part2);
  ln_finalize_kernel<<<BB, 32, 0, stream>>>(part2, stats2, G, invCnt);
  xi_kernel<<<(BB * NN * DD) / 256, 256, 0, stream>>>(xl, stats2, xi);
  wcvt_kernel<<<(DD * DD) / 256, 256, 0, stream>>>(wq, wk, wv, wo, wqT, wkT, wvT, woT);
  qkv_gemm_kernel<<<dim3(HH, (BB * NN) / 128, 3), 256, 0, stream>>>(xi, wqT, wkT, wvT, qb, kb, vb);
  attn_kernel<<<dim3(NN / 64, HH, BB), 128, 0, stream>>>(qb, kb, vb, ocat);
  out_proj_kernel<<<dim3(DD / 64, (BB * NN) / 128), 256, 0, stream>>>(ocat, woT, bo, xl, out);
}
